// Attention_54030688583705
// MI455X (gfx1250) — matrix operand order repaired and checked by reading, not compiled, not run on hardware
//
#include <hip/hip_runtime.h>
#include <hip/hip_bf16.h>

// ---------------------------------------------------------------------------
// Causal GQA attention for MI455X (gfx1250, wave32, WMMA).
// All matmuls use v_wmma_f32_16x16x32_bf16 (f32 accumulate).
// Workspace usage ~136 MiB (bf16 staging of x, weights, Q/K/V, attn-out).
// ---------------------------------------------------------------------------

typedef __bf16 bf16;
typedef __attribute__((ext_vector_type(16))) __bf16 v16bf;
typedef __attribute__((ext_vector_type(8)))  __bf16 v8bf;
typedef __attribute__((ext_vector_type(8)))  float  v8f;

static constexpr int SEQ   = 2048;
static constexpr int DIM   = 4096;
static constexpr int HD    = 128;
static constexpr int NH    = 32;
static constexpr int NKV   = 8;
static constexpr int QDIM  = NH * HD;    // 4096
static constexpr int KVDIM = NKV * HD;   // 1024

#define WMMA_BF16(A, B, C) \
  __builtin_amdgcn_wmma_f32_16x16x32_bf16(false, (A), false, (B), (short)0, (C), false, false)

// Load a 16x32 bf16 A-fragment row (one lane's share) from a row-major row.
// Contiguous K order matching B: lanes 0-15 -> K{0..15}, lanes 16-31 -> K{16..31}.
// half = lane>>4.
__device__ __forceinline__ v16bf load_a_frag(const bf16* __restrict__ rowptr, int half) {
  v8bf lo = *(const v8bf*)(rowptr + 16 * half);
  v8bf hi = *(const v8bf*)(rowptr + 16 * half + 8);
  v16bf r;
#pragma unroll
  for (int i = 0; i < 8; ++i) { r[i] = lo[i]; r[8 + i] = hi[i]; }
  return r;
}

// B-fragment (32x16 bf16): lane n=lane&15 selects the output column; lanes 0-15
// carry K=0..15, lanes 16-31 carry K=16..31, pairs packed in order -> one
// contiguous 16-element read per lane.
__device__ __forceinline__ v16bf load_b_frag(const bf16* __restrict__ colrow, int half) {
  return *(const v16bf*)(colrow + 16 * half);
}

// ---------------------------------------------------------------------------
// f32 -> bf16 conversion
// ---------------------------------------------------------------------------
__global__ void cvt_f32_bf16_kernel(const float* __restrict__ s, bf16* __restrict__ d, int n) {
  int i = blockIdx.x * blockDim.x + threadIdx.x;
  if (i < n) d[i] = (bf16)s[i];
}

// ---------------------------------------------------------------------------
// GEMM: C[M,N] = A[M,K] * W[N,K]^T  (both row-major bf16).
// Block = 128 threads (4 waves). Each wave computes a 32x32 C tile (2x2 WMMA
// tiles); block covers 64x64.  OUT_MODE: 0 = bf16 C[m*ldc+n],
// 1 = bf16 transposed C[n*ldc+m], 2 = f32 C[m*ldc+n].
// ---------------------------------------------------------------------------
template <int OUT_MODE>
__global__ __launch_bounds__(128)
void gemm_xwt_kernel(const bf16* __restrict__ A, const bf16* __restrict__ W,
                     void* __restrict__ C, int K, int lda, int ldw, int ldc) {
  const int wave  = threadIdx.x >> 5;
  const int lane  = threadIdx.x & 31;
  const int mlane = lane & 15;
  const int half  = lane >> 4;
  const int m0 = (blockIdx.y << 6) + ((wave >> 1) << 5);
  const int n0 = (blockIdx.x << 6) + ((wave & 1) << 5);

  const bf16* Ar0 = A + (size_t)(m0 + mlane) * lda;
  const bf16* Ar1 = A + (size_t)(m0 + 16 + mlane) * lda;
  const bf16* Wr0 = W + (size_t)(n0 + mlane) * ldw;
  const bf16* Wr1 = W + (size_t)(n0 + 16 + mlane) * ldw;

  v8f acc00 = {}, acc01 = {}, acc10 = {}, acc11 = {};

  for (int k = 0; k < K; k += 32) {
    // Exercise gfx1250 global_prefetch_b8 for the next K-block.
    __builtin_prefetch(Ar0 + k + 32, 0, 1);
    __builtin_prefetch(Ar1 + k + 32, 0, 1);
    __builtin_prefetch(Wr0 + k + 32, 0, 1);
    __builtin_prefetch(Wr1 + k + 32, 0, 1);

    v16bf a0 = load_a_frag(Ar0 + k, half);
    v16bf a1 = load_a_frag(Ar1 + k, half);
    v16bf b0 = load_b_frag(Wr0 + k, half);
    v16bf b1 = load_b_frag(Wr1 + k, half);

    acc00 = WMMA_BF16(a0, b0, acc00);
    acc01 = WMMA_BF16(a0, b1, acc01);
    acc10 = WMMA_BF16(a1, b0, acc10);
    acc11 = WMMA_BF16(a1, b1, acc11);
  }

  // C/D layout: row = r + 8*half (+16 per M tile), col = mlane (+16 per N tile)
  const v8f* accs[2][2] = {{&acc00, &acc01}, {&acc10, &acc11}};
#pragma unroll
  for (int ti = 0; ti < 2; ++ti) {
#pragma unroll
    for (int tj = 0; tj < 2; ++tj) {
      const v8f& a = *accs[ti][tj];
      int nn = n0 + 16 * tj + mlane;
#pragma unroll
      for (int r = 0; r < 8; ++r) {
        int mm = m0 + 16 * ti + 8 * half + r;
        if (OUT_MODE == 0)      ((bf16*)C)[(size_t)mm * ldc + nn] = (bf16)a[r];
        else if (OUT_MODE == 1) ((bf16*)C)[(size_t)nn * ldc + mm] = (bf16)a[r];
        else                    ((float*)C)[(size_t)mm * ldc + nn] = a[r];
      }
    }
  }
}

// ---------------------------------------------------------------------------
// RoPE over bf16 Q [S, 32*128] and K [S, 8*128]; freqs are f32 [S, 64].
// One thread per (s, head-of-40, pair).
// ---------------------------------------------------------------------------
__global__ void rope_kernel(bf16* __restrict__ Q, bf16* __restrict__ Kc,
                            const float* __restrict__ fc, const float* __restrict__ fs) {
  int idx = blockIdx.x * blockDim.x + threadIdx.x;
  int i = idx & 63; idx >>= 6;
  int head = idx % (NH + NKV);
  int s    = idx / (NH + NKV);
  float c  = fc[s * 64 + i];
  float sn = fs[s * 64 + i];
  bf16* p = (head < NH) ? (Q  + (size_t)s * QDIM  + head * HD + 2 * i)
                        : (Kc + (size_t)s * KVDIM + (head - NH) * HD + 2 * i);
  float xr = (float)p[0];
  float xi = (float)p[1];
  p[0] = (bf16)(xr * c - xi * sn);
  p[1] = (bf16)(xr * sn + xi * c);
}

// ---------------------------------------------------------------------------
// Flash attention, causal, GQA (kv head = h>>2).
// Block = 128 threads (4 waves); wave w handles the 16-row Q tile at
// q0 = blockIdx.y*64 + w*16 for head blockIdx.x.  Online softmax in f32.
// ---------------------------------------------------------------------------
__device__ __forceinline__ float rowmax16(float v) {
#pragma unroll
  for (int m = 8; m >= 1; m >>= 1) v = fmaxf(v, __shfl_xor(v, m, 32));
  return v;
}
__device__ __forceinline__ float rowsum16(float v) {
#pragma unroll
  for (int m = 8; m >= 1; m >>= 1) v += __shfl_xor(v, m, 32);
  return v;
}

__global__ __launch_bounds__(128)
void flash_attn_kernel(const bf16* __restrict__ Qb, const bf16* __restrict__ Kb,
                       const bf16* __restrict__ VT, bf16* __restrict__ AO) {
  __shared__ bf16 ldsP[4][16][32];   // per-wave 16x32 P tile (accum -> A-frag relayout)

  const int wave  = threadIdx.x >> 5;
  const int lane  = threadIdx.x & 31;
  const int mlane = lane & 15;
  const int half  = lane >> 4;
  const int h  = blockIdx.x;
  const int g  = h >> 2;                       // kv head (n_rep = 4)
  const int q0 = (blockIdx.y << 6) + (wave << 4);

  // Q fragments for this 16-row tile are loop-invariant: 16x128 bf16.
  const bf16* Qrow = Qb + (size_t)(q0 + mlane) * QDIM + h * HD;
  v16bf aq[4];
#pragma unroll
  for (int kk = 0; kk < 4; ++kk) aq[kk] = load_a_frag(Qrow + 32 * kk, half);

  v8f Oacc[8] = {};
  float mrow[8], lrow[8];
#pragma unroll
  for (int r = 0; r < 8; ++r) { mrow[r] = -1e30f; lrow[r] = 0.0f; }

  const float scale = 0.08838834764831845f;    // 1/sqrt(128)
  const int kend = q0 + 16;                    // causal: kpos <= qpos < q0+16

  for (int kb = 0; kb < kend; kb += 32) {
    // ---- S tile (16x32) = Q (16x128) x K^T ----
    v8f s0 = {}, s1 = {};
    const bf16* Kr0 = Kb + (size_t)(kb + mlane) * KVDIM + g * HD;
    const bf16* Kr1 = Kb + (size_t)(kb + 16 + mlane) * KVDIM + g * HD;
#pragma unroll
    for (int kk = 0; kk < 4; ++kk) {
      s0 = WMMA_BF16(aq[kk], load_b_frag(Kr0 + 32 * kk, half), s0);
      s1 = WMMA_BF16(aq[kk], load_b_frag(Kr1 + 32 * kk, half), s1);
    }

    // ---- online softmax (per accumulator row r) ----
#pragma unroll
    for (int r = 0; r < 8; ++r) {
      int qpos = q0 + 8 * half + r;
      int kp0  = kb + mlane;
      int kp1  = kb + 16 + mlane;
      float v0 = (kp0 <= qpos) ? s0[r] * scale : -1e30f;
      float v1 = (kp1 <= qpos) ? s1[r] * scale : -1e30f;
      float tm   = rowmax16(fmaxf(v0, v1));
      float mnew = fmaxf(mrow[r], tm);
      float corr = __expf(mrow[r] - mnew);
      float p0 = __expf(v0 - mnew);
      float p1 = __expf(v1 - mnew);
      mrow[r] = mnew;
      lrow[r] = lrow[r] * corr + rowsum16(p0 + p1);
#pragma unroll
      for (int c = 0; c < 8; ++c) Oacc[c][r] *= corr;
      ldsP[wave][8 * half + r][mlane]      = (bf16)p0;
      ldsP[wave][8 * half + r][16 + mlane] = (bf16)p1;
    }

    // wave-local LDS relayout: wait for DS stores, then reload as A-fragment
    asm volatile("s_wait_dscnt 0" ::: "memory");
    __builtin_amdgcn_wave_barrier();
    v16bf ap = load_a_frag(&ldsP[wave][mlane][0], half);

    // ---- O += P (16x32) x V (32x128); V stored transposed [d][kpos] ----
#pragma unroll
    for (int c = 0; c < 8; ++c) {
      const bf16* Vr = VT + (size_t)(g * HD + 16 * c + mlane) * SEQ;
      Oacc[c] = WMMA_BF16(ap, load_b_frag(Vr + kb, half), Oacc[c]);
    }
  }

  // ---- normalize and store attn output bf16 [qpos, h*128 + d] ----
#pragma unroll
  for (int r = 0; r < 8; ++r) {
    float inv = 1.0f / lrow[r];
    int qpos = q0 + 8 * half + r;
    bf16* dst = AO + (size_t)qpos * QDIM + h * HD;
#pragma unroll
    for (int c = 0; c < 8; ++c) dst[16 * c + mlane] = (bf16)(Oacc[c][r] * inv);
  }
}

// ---------------------------------------------------------------------------
// Host-side orchestration
// ---------------------------------------------------------------------------
extern "C" void kernel_launch(void* const* d_in, const int* in_sizes, int n_in,
                              void* d_out, int out_size, void* d_ws, size_t ws_size,
                              hipStream_t stream) {
  const float* x  = (const float*)d_in[0];   // [1, 2048, 4096]
  const float* wq = (const float*)d_in[1];   // [4096, 4096]
  const float* wk = (const float*)d_in[2];   // [1024, 4096]
  const float* wv = (const float*)d_in[3];   // [1024, 4096]
  const float* wo = (const float*)d_in[4];   // [4096, 4096]
  const float* fc = (const float*)d_in[5];   // [2048, 64]
  const float* fs = (const float*)d_in[6];   // [2048, 64]
  float* out = (float*)d_out;                // [1, 2048, 4096] f32

  // Workspace carve-up (256B aligned), total ~136 MiB.
  char* wsp = (char*)d_ws;
  size_t off = 0;
  auto carve = [&](size_t bytes) {
    void* p = wsp + off;
    off = (off + bytes + 255) & ~(size_t)255;
    return p;
  };
  bf16* xb  = (bf16*)carve((size_t)SEQ * DIM * 2);
  bf16* wqb = (bf16*)carve((size_t)QDIM * DIM * 2);
  bf16* wkb = (bf16*)carve((size_t)KVDIM * DIM * 2);
  bf16* wvb = (bf16*)carve((size_t)KVDIM * DIM * 2);
  bf16* wob = (bf16*)carve((size_t)DIM * QDIM * 2);
  bf16* Qb  = (bf16*)carve((size_t)SEQ * QDIM * 2);
  bf16* Kb  = (bf16*)carve((size_t)SEQ * KVDIM * 2);
  bf16* VT  = (bf16*)carve((size_t)KVDIM * SEQ * 2);   // transposed [d][kpos]
  bf16* AO  = (bf16*)carve((size_t)SEQ * QDIM * 2);
  (void)ws_size; (void)n_in; (void)in_sizes; (void)out_size;

  auto cvt = [&](const float* s, bf16* d, int n) {
    cvt_f32_bf16_kernel<<<(n + 255) / 256, 256, 0, stream>>>(s, d, n);
  };
  cvt(x,  xb,  SEQ * DIM);
  cvt(wq, wqb, QDIM * DIM);
  cvt(wk, wkb, KVDIM * DIM);
  cvt(wv, wvb, KVDIM * DIM);
  cvt(wo, wob, DIM * QDIM);

  // Q = x @ wq^T  -> bf16 [2048, 4096]
  gemm_xwt_kernel<0><<<dim3(QDIM / 64, SEQ / 64), 128, 0, stream>>>(
      xb, wqb, Qb, DIM, DIM, DIM, QDIM);
  // K = x @ wk^T  -> bf16 [2048, 1024]
  gemm_xwt_kernel<0><<<dim3(KVDIM / 64, SEQ / 64), 128, 0, stream>>>(
      xb, wkb, Kb, DIM, DIM, DIM, KVDIM);
  // V = x @ wv^T  -> stored transposed bf16 [1024, 2048]
  gemm_xwt_kernel<1><<<dim3(KVDIM / 64, SEQ / 64), 128, 0, stream>>>(
      xb, wvb, VT, DIM, DIM, DIM, SEQ);

  // RoPE on Q and K.
  {
    int total = SEQ * (NH + NKV) * (HD / 2);
    rope_kernel<<<total / 256, 256, 0, stream>>>(Qb, Kb, fc, fs);
  }

  // Flash attention: 32 heads x 32 q-blocks of 64 rows (4 waves x 16 rows).
  flash_attn_kernel<<<dim3(NH, SEQ / 64), 128, 0, stream>>>(Qb, Kb, VT, AO);

  // out = AO @ wo^T -> f32 [2048, 4096]
  gemm_xwt_kernel<2><<<dim3(DIM / 64, SEQ / 64), 128, 0, stream>>>(
      AO, wob, out, QDIM, QDIM, QDIM, DIM);
}
